// GQATrainingAttention_58222576664985
// MI455X (gfx1250) — matrix operand order repaired and checked by reading, not compiled, not run on hardware
//
#include <hip/hip_runtime.h>
#include <hip/hip_bf16.h>
#include <stdint.h>

// ---- problem constants (match reference) ----
#define TDIM   2048
#define NHEAD  16
#define NKV    4
#define HDIM   128
#define DMODEL 2048   // H*HD
#define DKV    512    // KV*HD
#define BATCH  2

typedef __attribute__((ext_vector_type(16))) __bf16 v16bf;
typedef __attribute__((ext_vector_type(8)))  float  v8f;
typedef __attribute__((ext_vector_type(4)))  unsigned int u32x4;
typedef __attribute__((ext_vector_type(8)))  int          i32x8;
typedef __attribute__((ext_vector_type(4)))  int          i32x4;

union AF { v16bf v; unsigned int u[8]; };
union CF { v8f   v; float        f[8]; };

#if defined(__has_builtin)
#if __has_builtin(__builtin_amdgcn_tensor_load_to_lds)
#define HAVE_TDM 1
#endif
#endif
#ifndef HAVE_TDM
#define HAVE_TDM 0
#endif

__device__ __forceinline__ unsigned short f2bf(float f) {
  union { float f; unsigned int u; } x; x.f = f;
  unsigned int u = x.u;
  unsigned int r = u + 0x7FFFu + ((u >> 16) & 1u);   // round-to-nearest-even
  return (unsigned short)(r >> 16);
}

// ---------------------------------------------------------------------------
// One-shot fp32 -> bf16 conversion (hoists conversion VALU out of GEMM loops)
// ---------------------------------------------------------------------------
__global__ __launch_bounds__(256)
void cvt_bf16(const float* __restrict__ src, unsigned short* __restrict__ dst, int n)
{
  int idx = blockIdx.x * blockDim.x + threadIdx.x;
  int stride = gridDim.x * blockDim.x;
  for (int i = idx * 4; i < n; i += stride * 4) {
    float4 v = *(const float4*)(src + i);
    unsigned int lo = (unsigned)f2bf(v.x) | ((unsigned)f2bf(v.y) << 16);
    unsigned int hi = (unsigned)f2bf(v.z) | ((unsigned)f2bf(v.w) << 16);
    *(uint2*)(dst + i) = make_uint2(lo, hi);
  }
}

// ---------------------------------------------------------------------------
// TDM: issue a 2D tile load (tile_dim0 = 32 bf16 elems, tile_dim1 = 128 rows)
// into LDS. Descriptor built per CDNA5 ISA D# group0/group1 layout.
// This toolchain exposes the 6-arg builtin:
//   (u32x4 g0, i32x8 g1, i32x4 g2, i32x4 g3, i32x8 gx, i32 cpol)
// ---------------------------------------------------------------------------
#if HAVE_TDM
__device__ __forceinline__ void tdm_load_a_tile(const unsigned short* gtile,
                                                unsigned lds_byte_addr, int K)
{
  unsigned long long ga = (unsigned long long)(uintptr_t)gtile;
  u32x4 g0;
  g0[0] = 1u;                                              // count=1, user mode
  g0[1] = lds_byte_addr;                                   // lds_addr
  g0[2] = (unsigned)(ga & 0xffffffffull);                  // global_addr[31:0]
  g0[3] = (unsigned)((ga >> 32) & 0x01ffffffull)           // global_addr[56:32]
        | 0x80000000u;                                     // type=2 ("image")
  unsigned td0 = (unsigned)K;                              // tensor_dim0
  unsigned td1 = 0x00ffffffu;                              // tensor_dim1 (large)
  i32x8 g1;
  g1[0] = (int)(1u << 16);                                 // data_size = 2 bytes
  g1[1] = (int)((td0 & 0xffffu) << 16);                    // tensor_dim0[15:0]
  g1[2] = (int)((td0 >> 16) | ((td1 & 0xffffu) << 16));    // td0 hi | td1 lo
  g1[3] = (int)((td1 >> 16) | (32u << 16));                // td1 hi | tile_dim0=32
  g1[4] = (int)(128u);                                     // tile_dim1=128, tile_dim2=0
  g1[5] = (int)((unsigned)K);                              // tensor_dim0_stride lo
  g1[6] = 0;                                               // stride0 hi | stride1 lo
  g1[7] = 0;                                               // stride1 hi
  i32x4 z4 = {0, 0, 0, 0};                                 // groups 2/3 unused (2D)
  i32x8 z8 = {0, 0, 0, 0, 0, 0, 0, 0};
  __builtin_amdgcn_tensor_load_to_lds(g0, g1, z4, z4, z8, 0);
}
#endif

// ---------------------------------------------------------------------------
// bf16 GEMM: C[M,N] = A[M,K] * B[K,N], A/B bf16, C bf16 or f32.
// Block tile 128x128, K-step 32, 256 threads = 8 waves (4x2 wave grid, each
// wave 32x64 = 8 accumulators). Double-buffered LDS; A tile staged by the
// Tensor Data Mover (wave 0 issues, s_wait_tensorcnt + barrier publishes),
// B tile staged transposed ([n][k]) by all threads.
// ---------------------------------------------------------------------------
template<bool OUT_BF16>
__global__ __launch_bounds__(256)
void gemm_bf16_wmma(const unsigned short* __restrict__ Ab,
                    const unsigned short* __restrict__ Bb,
                    int M, int N, int K, void* __restrict__ Cp)
{
  __shared__ unsigned short Alds[2][128][32];   // [buf][m][k]
  __shared__ unsigned short Blds[2][128][32];   // [buf][n][k]

  const int tid  = threadIdx.x;
  const int wave = tid >> 5;
  const int lane = tid & 31;
  const int hi   = lane >> 4;
  const int ln   = lane & 15;
  const int wm   = (wave >> 1) * 32;      // wave m-origin inside block
  const int wn   = (wave & 1) * 64;       // wave n-origin inside block
  const int m_blk = blockIdx.y * 128;
  const int n_blk = blockIdx.x * 128;

  v8f zero = {0.f,0.f,0.f,0.f,0.f,0.f,0.f,0.f};
  CF acc[2][4];
#pragma unroll
  for (int mt = 0; mt < 2; ++mt)
#pragma unroll
    for (int nt = 0; nt < 4; ++nt) acc[mt][nt].v = zero;

  const int nk = K >> 5;

  auto load_a = [&](int kk, int buf) {
#if HAVE_TDM
    if (wave == 0)
      tdm_load_a_tile(Ab + (size_t)m_blk * (size_t)K + (size_t)(kk * 32),
                      (unsigned)(uintptr_t)&Alds[buf][0][0], K);
#else
#pragma unroll 2
    for (int i = tid; i < 128 * 16; i += 256) {
      int r = i >> 4, c = i & 15;
      ((unsigned int*)&Alds[buf][r][0])[c] =
          ((const unsigned int*)(Ab + (size_t)(m_blk + r) * (size_t)K + (size_t)(kk * 32)))[c];
    }
#endif
  };

  auto load_b = [&](int kk, int buf) {
    // 32 k-rows x 128 n-cols, transpose into [n][k]
#pragma unroll 2
    for (int i = tid; i < 32 * 64; i += 256) {
      int r = i >> 6, c = i & 63;   // r = k row, c = u32 (pair of n)
      unsigned int w =
          ((const unsigned int*)(Bb + (size_t)(kk * 32 + r) * (size_t)N + (size_t)n_blk))[c];
      Blds[buf][2 * c][r]     = (unsigned short)(w & 0xffffu);
      Blds[buf][2 * c + 1][r] = (unsigned short)(w >> 16);
    }
  };

  load_a(0, 0);
  load_b(0, 0);

  for (int kk = 0; kk < nk; ++kk) {
    const int buf = kk & 1;
#if HAVE_TDM
    if (wave == 0) __builtin_amdgcn_s_wait_tensorcnt(0);
#endif
    __syncthreads();                       // publish tile kk (TDM + ds stores)

    if (kk + 1 < nk) {                     // prefetch next tile into other buffer
      load_a(kk + 1, buf ^ 1);
      load_b(kk + 1, buf ^ 1);
    }

    AF a[2];
#pragma unroll
    for (int mt = 0; mt < 2; ++mt) {
      const unsigned int* Arow = (const unsigned int*)(&Alds[buf][wm + mt * 16 + ln][0]);
#pragma unroll
      for (int v = 0; v < 8; ++v) a[mt].u[v] = Arow[v + 8 * hi];
    }
#pragma unroll
    for (int nt = 0; nt < 4; ++nt) {
      AF bf;
      const unsigned int* Brow = (const unsigned int*)(&Blds[buf][wn + nt * 16 + ln][0]);
#pragma unroll
      for (int v = 0; v < 8; ++v) bf.u[v] = Brow[v + 8 * hi];
#pragma unroll
      for (int mt = 0; mt < 2; ++mt)
        acc[mt][nt].v = __builtin_amdgcn_wmma_f32_16x16x32_bf16(
            false, a[mt].v, false, bf.v, (short)0, acc[mt][nt].v, false, false);
    }
  }

#pragma unroll
  for (int mt = 0; mt < 2; ++mt)
#pragma unroll
    for (int nt = 0; nt < 4; ++nt)
#pragma unroll
      for (int r = 0; r < 8; ++r) {
        size_t row = (size_t)(m_blk + wm + mt * 16 + r + 8 * hi);
        size_t col = (size_t)(n_blk + wn + nt * 16 + ln);
        float val = acc[mt][nt].f[r];
        if constexpr (OUT_BF16) ((unsigned short*)Cp)[row * (size_t)N + col] = f2bf(val);
        else                    ((float*)Cp)[row * (size_t)N + col] = val;
      }
}

// ---------------------------------------------------------------------------
// Flash attention: 4 waves / 64 queries per block; waves share the K/V LDS
// stage (4x traffic amortization). Block-uniform causal key loop; lower waves
// see fully masked extra key blocks (p == 0 -> exact no-op).
// ---------------------------------------------------------------------------
__global__ __launch_bounds__(128)
void attn_wmma(const unsigned short* __restrict__ Qb,
               const unsigned short* __restrict__ Kb,
               const unsigned short* __restrict__ Vb,
               const int* __restrict__ ids,
               unsigned short* __restrict__ AO)
{
  __shared__ unsigned short q_lds[64][128];     // [q][hd]
  __shared__ unsigned short k_lds[32][128];     // [key][hd]  (B layout for QK^T)
  __shared__ unsigned short vt_lds[128][34];    // [hd][key]  (B layout for P@V)
  __shared__ unsigned short p_lds[4][16][32];   // per-wave [q][key] (A layout)

  const int tid  = threadIdx.x;
  const int wv   = tid >> 5;
  const int lane = tid & 31;
  const int hi = lane >> 4, ln = lane & 15;
  const int q_blk = blockIdx.x * 64;
  const int q0 = q_blk + wv * 16;
  const int h  = blockIdx.y;
  const int b  = blockIdx.z;
  const int kvh = h & (NKV - 1);                // head h -> kv head h % KV

  // ---- load 64x128 Q tile ----
  for (int i = tid; i < 64 * 64; i += 128) {
    int r = i >> 6, c = i & 63;
    const unsigned int* s =
        (const unsigned int*)(Qb + ((size_t)(b * TDIM + q_blk + r)) * DMODEL + h * HDIM);
    ((unsigned int*)q_lds)[i] = s[c];
  }
  __syncthreads();

  AF qa[4];
  {
    const unsigned int* Arow = (const unsigned int*)(&q_lds[wv * 16 + ln][0]);
#pragma unroll
    for (int kc = 0; kc < 4; ++kc)
#pragma unroll
      for (int v = 0; v < 8; ++v)
        qa[kc].u[v] = Arow[v + 8 * hi + 16 * kc];
  }

  v8f zero = {0.f,0.f,0.f,0.f,0.f,0.f,0.f,0.f};
  float row_max[8], row_sum[8];
  CF o[8];
#pragma unroll
  for (int r = 0; r < 8; ++r) { row_max[r] = -1e30f; row_sum[r] = 0.f; }
#pragma unroll
  for (int nt = 0; nt < 8; ++nt) o[nt].v = zero;

  const float scale = 0.08838834764831845f;     // 1/sqrt(128)
  const int k_last = q_blk + 63;                // block-uniform causal bound

  for (int k0 = 0; k0 <= k_last; k0 += 32) {
    __syncthreads();
    // K tile (32 x 128) -> [key][hd]
    for (int i = tid; i < 32 * 64; i += 128) {
      int r = i >> 6, c = i & 63;
      const unsigned int* s =
          (const unsigned int*)(Kb + ((size_t)(b * TDIM + k0 + r)) * DKV + kvh * HDIM);
      ((unsigned int*)k_lds)[i] = s[c];
    }
    // V tile transposed -> [hd][key]
    for (int i = tid; i < 32 * 64; i += 128) {
      int r = i >> 6, c = i & 63;
      const unsigned int* s =
          (const unsigned int*)(Vb + ((size_t)(b * TDIM + k0 + r)) * DKV + kvh * HDIM);
      unsigned int w = s[c];
      vt_lds[2 * c][r]     = (unsigned short)(w & 0xffffu);
      vt_lds[2 * c + 1][r] = (unsigned short)(w >> 16);
    }
    __syncthreads();

    // ---- scores: two 16x16 tiles over hd=128 ----
    CF sc[2];
#pragma unroll
    for (int st = 0; st < 2; ++st) {
      sc[st].v = zero;
      const unsigned int* Brow = (const unsigned int*)(&k_lds[st * 16 + ln][0]);
#pragma unroll
      for (int kc = 0; kc < 4; ++kc) {
        AF kb;
#pragma unroll
        for (int v = 0; v < 8; ++v) kb.u[v] = Brow[v + 8 * hi + 16 * kc];
        sc[st].v = __builtin_amdgcn_wmma_f32_16x16x32_bf16(
            false, qa[kc].v, false, kb.v, (short)0, sc[st].v, false, false);
      }
    }

    // ---- scale + causal & padding mask ----
#pragma unroll
    for (int st = 0; st < 2; ++st) {
      int j = k0 + st * 16 + ln;
      bool pad_ok = (ids[b * TDIM + j] != 0);
#pragma unroll
      for (int r = 0; r < 8; ++r) {
        int qi = q0 + r + 8 * hi;
        bool ok = pad_ok && (j <= qi);
        sc[st].f[r] = ok ? sc[st].f[r] * scale : -1e30f;
      }
    }

    // ---- online softmax (row r+8*hi lives in one 16-lane half) ----
    float alpha[8];
#pragma unroll
    for (int r = 0; r < 8; ++r) {
      float bm = fmaxf(sc[0].f[r], sc[1].f[r]);
#pragma unroll
      for (int msk = 1; msk < 16; msk <<= 1)
        bm = fmaxf(bm, __shfl_xor(bm, msk, 32));
      float nm = fmaxf(row_max[r], bm);
      alpha[r] = __expf(row_max[r] - nm);
      row_max[r] = nm;
      float p0 = __expf(sc[0].f[r] - nm);
      float p1 = __expf(sc[1].f[r] - nm);
      sc[0].f[r] = p0; sc[1].f[r] = p1;
      float ps = p0 + p1;
#pragma unroll
      for (int msk = 1; msk < 16; msk <<= 1)
        ps += __shfl_xor(ps, msk, 32);
      row_sum[r] = row_sum[r] * alpha[r] + ps;
    }

    // ---- P to per-wave LDS as bf16 (C-frag layout -> A-frag layout) ----
#pragma unroll
    for (int st = 0; st < 2; ++st)
#pragma unroll
      for (int r = 0; r < 8; ++r)
        p_lds[wv][r + 8 * hi][st * 16 + ln] = f2bf(sc[st].f[r]);
    __syncthreads();

#pragma unroll
    for (int nt = 0; nt < 8; ++nt)
#pragma unroll
      for (int r = 0; r < 8; ++r)
        o[nt].f[r] *= alpha[r];

    AF pa;
    const unsigned int* Prow = (const unsigned int*)(&p_lds[wv][ln][0]);
#pragma unroll
    for (int v = 0; v < 8; ++v) pa.u[v] = Prow[v + 8 * hi];

#pragma unroll
    for (int nt = 0; nt < 8; ++nt) {
      AF vb;
      const unsigned int* Vrow = (const unsigned int*)(&vt_lds[nt * 16 + ln][0]);
#pragma unroll
      for (int v = 0; v < 8; ++v) vb.u[v] = Vrow[v + 8 * hi];
      o[nt].v = __builtin_amdgcn_wmma_f32_16x16x32_bf16(
          false, pa.v, false, vb.v, (short)0, o[nt].v, false, false);
    }
  }

  // ---- normalize and store (bf16, [B*T, DMODEL]) ----
#pragma unroll
  for (int r = 0; r < 8; ++r) row_sum[r] = 1.0f / row_sum[r];
#pragma unroll
  for (int nt = 0; nt < 8; ++nt)
#pragma unroll
    for (int r = 0; r < 8; ++r) {
      size_t row = (size_t)(b * TDIM + q0 + r + 8 * hi);
      size_t col = (size_t)(h * HDIM + nt * 16 + ln);
      AO[row * DMODEL + col] = f2bf(o[nt].f[r] * row_sum[r]);
    }
}

// ---------------------------------------------------------------------------
extern "C" void kernel_launch(void* const* d_in, const int* in_sizes, int n_in,
                              void* d_out, int out_size, void* d_ws, size_t ws_size,
                              hipStream_t stream) {
  (void)in_sizes; (void)n_in; (void)out_size; (void)ws_size;
  const float* X   = (const float*)d_in[0];
  const int*   ids = (const int*)d_in[1];
  const float* Wq  = (const float*)d_in[2];
  const float* Wk  = (const float*)d_in[3];
  const float* Wv  = (const float*)d_in[4];
  const float* Wo  = (const float*)d_in[5];

  const int M = BATCH * TDIM;   // 4096

  unsigned short* Xb  = (unsigned short*)d_ws;
  unsigned short* Wqb = Xb  + (size_t)M * DMODEL;
  unsigned short* Wkb = Wqb + (size_t)DMODEL * DMODEL;
  unsigned short* Wvb = Wkb + (size_t)DMODEL * DKV;
  unsigned short* Wob = Wvb + (size_t)DMODEL * DKV;
  unsigned short* Qb  = Wob + (size_t)DMODEL * DMODEL;
  unsigned short* Kb  = Qb  + (size_t)M * DMODEL;
  unsigned short* Vb  = Kb  + (size_t)M * DKV;
  unsigned short* AO  = Vb  + (size_t)M * DKV;

  cvt_bf16<<<1024, 256, 0, stream>>>(X,  Xb,  M * DMODEL);
  cvt_bf16<<<1024, 256, 0, stream>>>(Wq, Wqb, DMODEL * DMODEL);
  cvt_bf16<<< 256, 256, 0, stream>>>(Wk, Wkb, DMODEL * DKV);
  cvt_bf16<<< 256, 256, 0, stream>>>(Wv, Wvb, DMODEL * DKV);
  cvt_bf16<<<1024, 256, 0, stream>>>(Wo, Wob, DMODEL * DMODEL);

  gemm_bf16_wmma<true><<<dim3(DMODEL / 128, M / 128), 256, 0, stream>>>(Xb, Wqb, M, DMODEL, DMODEL, Qb);
  gemm_bf16_wmma<true><<<dim3(DKV    / 128, M / 128), 256, 0, stream>>>(Xb, Wkb, M, DKV,    DMODEL, Kb);
  gemm_bf16_wmma<true><<<dim3(DKV    / 128, M / 128), 256, 0, stream>>>(Xb, Wvb, M, DKV,    DMODEL, Vb);

  attn_wmma<<<dim3(TDIM / 64, NHEAD, BATCH), 128, 0, stream>>>(Qb, Kb, Vb, ids, AO);

  gemm_bf16_wmma<false><<<dim3(DMODEL / 128, M / 128), 256, 0, stream>>>(AO, Wob, M, DMODEL, DMODEL, (float*)d_out);
}
